// RNNoiseGRU_26465588478585
// MI455X (gfx1250) — matrix operand order repaired and checked by reading, not compiled, not run on hardware
//
#include <hip/hip_runtime.h>
#include <hip/hip_bf16.h>

typedef __attribute__((ext_vector_type(16))) _Float16 v16h;
typedef __attribute__((ext_vector_type(8)))  float    v8f;
typedef __attribute__((ext_vector_type(4)))  float    v4f;

#define FRAME 480
#define HID   24
#define G3    72   // 3*HID

// ---------------------------------------------------------------------------
// Packed-weight layout in d_ws (all f16, 512 halves per 16x16x32 B-fragment):
//   frag 0..29  : W_in  B-frags, frag = kc*2 + nt   (kc=0..14, nt=0..1)
//   frag 30..44 : GRU   B-frags, frag = 30 + l*5+nt (l=0..2,  nt=0..4)
//   frag 45..74 : W_out B-frags, frag = 45 + nt     (nt=0..29)
// Within a fragment: lane L holds 16 halves at offset L*16.
// Assumed B layout (32x16, 16-bit): lane L -> column N = L%16 (+tile base),
//   K = (L/16)*8 + j (j<8), 16 + (L/16)*8 + (j-8) (j>=8), packed pairs per dword.
// ---------------------------------------------------------------------------
#define NFRAG_TOTAL 75

__global__ __launch_bounds__(256) void pack_weights_kernel(
    const float* __restrict__ W_in,   // [24,480]
    const float* __restrict__ W_out,  // [480,24]
    const float* __restrict__ Wih1,   // [72,24]
    const float* __restrict__ Wih2,
    const float* __restrict__ Wih3,
    _Float16* __restrict__ ws)
{
    int t = blockIdx.x * blockDim.x + threadIdx.x;
    if (t >= NFRAG_TOTAL * 512) return;
    int frag   = t >> 9;          // which fragment
    int lane_j = t & 511;
    int lane   = lane_j >> 4;
    int j      = lane_j & 15;
    int ln = lane & 15, hh = lane >> 4;

    float v = 0.0f;
    if (frag < 30) {                       // W_in (B = W_in^T): B[k][n] = W_in[n][k]
        int kc = frag >> 1, nt = frag & 1;
        int k = kc * 32 + (j < 8 ? hh * 8 + j : 16 + hh * 8 + (j - 8));     // k < 480 always
        int n = nt * 16 + ln;
        if (n < HID) v = W_in[n * FRAME + k];
    } else if (frag < 45) {                // GRU W_ih: B[k][n] = W_ih[n][k], K padded 24->32
        int f = frag - 30;
        int l = f / 5, nt = f % 5;
        int k = (j < 8 ? hh * 8 + j : 16 + hh * 8 + (j - 8));
        int n = nt * 16 + ln;
        const float* W = (l == 0) ? Wih1 : ((l == 1) ? Wih2 : Wih3);
        if (n < G3 && k < HID) v = W[n * HID + k];
    } else {                               // W_out (B = W_out^T): B[k][n] = W_out[n][k]
        int nt = frag - 45;
        int k = (j < 8 ? hh * 8 + j : 16 + hh * 8 + (j - 8));
        int n = nt * 16 + ln;              // n < 480 always
        if (k < HID) v = W_out[n * HID + k];
    }
    ws[t] = (_Float16)v;
}

union AF { v16h v; uint4 u[2]; };

__device__ __forceinline__ float fast_sigmoid(float x) {
    return 1.0f / (1.0f + __expf(-x));
}
__device__ __forceinline__ float fast_tanh(float x) {
    float e = __expf(2.0f * x);
    return (e - 1.0f) / (e + 1.0f);
}

// ---------------------------------------------------------------------------
// Main kernel: 8 waves/WG, each wave independently handles 16 rows.
// No barriers needed: per-wave LDS slices; LDS ops are in-order per wave.
// ---------------------------------------------------------------------------
__global__ __launch_bounds__(256) void rnnoise_gru_kernel(
    const float* __restrict__ x,
    const float* __restrict__ b_in,
    const float* __restrict__ b_out,
    const float* __restrict__ bih1, const float* __restrict__ bhh1,
    const float* __restrict__ bih2, const float* __restrict__ bhh2,
    const float* __restrict__ bih3, const float* __restrict__ bhh3,
    const _Float16* __restrict__ wpk,
    float* __restrict__ out)
{
    // Per-wave scratch: A-fragment buffer 16x40 halves, staging buffer 16x80 f32
    __shared__ alignas(16) _Float16 sA[8][16 * 40];
    __shared__ alignas(16) float    sG[8][16 * 80];

    const int tid  = threadIdx.x;
    const int wid  = tid >> 5;
    const int lane = tid & 31;
    const int ln   = lane & 15;
    const int hh   = lane >> 4;

    _Float16* Ab = sA[wid];
    float*    Gb = sG[wid];
    const long r0 = (long)blockIdx.x * 128 + (long)wid * 16;

    // ===== Phase 1: feat[16,24] = x[16,480] @ W_in^T, WMMA over 15 K-chunks =====
    v8f acc0 = {}; v8f acc1 = {};
    const float* xrow = x + (r0 + ln) * FRAME + hh * 8;
    #pragma unroll
    for (int kc = 0; kc < 15; ++kc) {
        const float* p = xrow + kc * 32;
        v4f p0 = *(const v4f*)(p);
        v4f p1 = *(const v4f*)(p + 4);
        v4f p2 = *(const v4f*)(p + 16);
        v4f p3 = *(const v4f*)(p + 20);
        v16h a;
        a[0]=(_Float16)p0[0]; a[1]=(_Float16)p0[1]; a[2]=(_Float16)p0[2]; a[3]=(_Float16)p0[3];
        a[4]=(_Float16)p1[0]; a[5]=(_Float16)p1[1]; a[6]=(_Float16)p1[2]; a[7]=(_Float16)p1[3];
        a[8]=(_Float16)p2[0]; a[9]=(_Float16)p2[1]; a[10]=(_Float16)p2[2]; a[11]=(_Float16)p2[3];
        a[12]=(_Float16)p3[0]; a[13]=(_Float16)p3[1]; a[14]=(_Float16)p3[2]; a[15]=(_Float16)p3[3];

        AF b0, b1;
        const _Float16* f0 = wpk + (size_t)(kc * 2 + 0) * 512 + lane * 16;
        const _Float16* f1 = wpk + (size_t)(kc * 2 + 1) * 512 + lane * 16;
        b0.u[0] = *(const uint4*)(f0); b0.u[1] = *(const uint4*)(f0 + 8);
        b1.u[0] = *(const uint4*)(f1); b1.u[1] = *(const uint4*)(f1 + 8);

        acc0 = __builtin_amdgcn_wmma_f32_16x16x32_f16(false, a, false, b0.v, (short)0, acc0, false, false);
        acc1 = __builtin_amdgcn_wmma_f32_16x16x32_f16(false, a, false, b1.v, (short)0, acc1, false, false);
    }
    // Store feat (+b_in) as f16 A-fragment layout into Ab; zero K-pad cols 24..31
    {
        float bi0 = b_in[ln];
        float bi1 = (ln < 8) ? b_in[16 + ln] : 0.0f;
        #pragma unroll
        for (int i = 0; i < 8; ++i) {
            int m = hh * 8 + i;   // D layout: vgpr i -> M = i (lanes<16) / 8+i (lanes>=16)
            Ab[m * 40 + ln]      = (_Float16)(acc0[i] + bi0);
            Ab[m * 40 + 16 + ln] = (ln < 8) ? (_Float16)(acc1[i] + bi1) : (_Float16)0.0f;
        }
    }

    // ===== Phase 2: three GRU steps (h0=0 -> W_hh vanishes, b_hh stays) =====
    #pragma unroll
    for (int l = 0; l < 3; ++l) {
        AF a;
        const _Float16* ap = Ab + ln * 40 + hh * 8;
        a.u[0] = *(const uint4*)(ap);
        a.u[1] = *(const uint4*)(ap + 16);
        #pragma unroll
        for (int t = 0; t < 5; ++t) {
            const _Float16* f = wpk + (size_t)(30 + l * 5 + t) * 512 + lane * 16;
            AF b;
            b.u[0] = *(const uint4*)(f); b.u[1] = *(const uint4*)(f + 8);
            v8f z = {};
            v8f acc = __builtin_amdgcn_wmma_f32_16x16x32_f16(false, a.v, false, b.v, (short)0, z, false, false);
            #pragma unroll
            for (int i = 0; i < 8; ++i)
                Gb[(hh * 8 + i) * 80 + t * 16 + ln] = acc[i];
        }
        const float* bih = (l == 0) ? bih1 : ((l == 1) ? bih2 : bih3);
        const float* bhh = (l == 0) ? bhh1 : ((l == 1) ? bhh2 : bhh3);
        #pragma unroll 4
        for (int m = 0; m < 16; ++m) {
            if (lane < HID) {
                int j = lane;
                float gr = Gb[m * 80 + j]          + bih[j]          + bhh[j];
                float gz = Gb[m * 80 + HID + j]    + bih[HID + j]    + bhh[HID + j];
                float gn = Gb[m * 80 + 2*HID + j]  + bih[2*HID + j];
                float r  = fast_sigmoid(gr);
                float zg = fast_sigmoid(gz);
                float nc = fast_tanh(gn + r * bhh[2*HID + j]);
                Ab[m * 40 + j] = (_Float16)((1.0f - zg) * nc);
            }
        }
    }

    // ===== Phase 3: gain = sigmoid(h3 @ W_out^T + b_out); out = x * gain =====
    AF a3;
    {
        const _Float16* ap = Ab + ln * 40 + hh * 8;
        a3.u[0] = *(const uint4*)(ap);
        a3.u[1] = *(const uint4*)(ap + 16);
    }
    for (int ch = 0; ch < 6; ++ch) {          // 6 chunks x 80 output columns
        #pragma unroll
        for (int t = 0; t < 5; ++t) {
            int nt = ch * 5 + t;
            const _Float16* f = wpk + (size_t)(45 + nt) * 512 + lane * 16;
            AF b;
            b.u[0] = *(const uint4*)(f); b.u[1] = *(const uint4*)(f + 8);
            v8f z = {};
            v8f acc = __builtin_amdgcn_wmma_f32_16x16x32_f16(false, a3.v, false, b.v, (short)0, z, false, false);
            float bo = b_out[nt * 16 + ln];
            #pragma unroll
            for (int i = 0; i < 8; ++i)
                Gb[(hh * 8 + i) * 80 + t * 16 + ln] = fast_sigmoid(acc[i] + bo);
        }
        // Coalesced read x / multiply / non-temporal store out (v4f per lane).
        // NT store: out is write-once, never re-read; keep x resident in the
        // 192MB L2 (whole x fits) so this phase's re-reads stay L2 hits.
        #pragma unroll
        for (int it = 0; it < 10; ++it) {
            int idx = it * 32 + lane;         // 0..319  (16 rows x 20 float4s)
            int rq  = idx / 20;
            int c4  = idx % 20;
            long gofs = (r0 + rq) * FRAME + ch * 80 + c4 * 4;
            v4f xv = *(const v4f*)(x + gofs);
            v4f gv = *(const v4f*)(Gb + rq * 80 + c4 * 4);
            v4f ov = xv * gv;
            __builtin_nontemporal_store(ov, (v4f*)(out + gofs));
        }
    }
}

extern "C" void kernel_launch(void* const* d_in, const int* in_sizes, int n_in,
                              void* d_out, int out_size, void* d_ws, size_t ws_size,
                              hipStream_t stream) {
    const float* x     = (const float*)d_in[0];
    const float* W_in  = (const float*)d_in[1];
    const float* b_in  = (const float*)d_in[2];
    const float* W_out = (const float*)d_in[3];
    const float* b_out = (const float*)d_in[4];
    const float* Wih1  = (const float*)d_in[5];
    const float* bih1  = (const float*)d_in[7];
    const float* bhh1  = (const float*)d_in[8];
    const float* Wih2  = (const float*)d_in[9];
    const float* bih2  = (const float*)d_in[11];
    const float* bhh2  = (const float*)d_in[12];
    const float* Wih3  = (const float*)d_in[13];
    const float* bih3  = (const float*)d_in[15];
    const float* bhh3  = (const float*)d_in[16];

    _Float16* wpk = (_Float16*)d_ws;

    // Pack weights into WMMA B-fragment layout (75 frags x 512 halves = 76.8 KB)
    int pack_threads = NFRAG_TOTAL * 512;
    pack_weights_kernel<<<(pack_threads + 255) / 256, 256, 0, stream>>>(
        W_in, W_out, Wih1, Wih2, Wih3, wpk);

    int B = in_sizes[0] / FRAME;          // 65536
    int grid = B / 128;                   // 128 rows per workgroup
    rnnoise_gru_kernel<<<grid, 256, 0, stream>>>(
        x, b_in, b_out, bih1, bhh1, bih2, bhh2, bih3, bhh3,
        wpk, (float*)d_out);
}
